// IntegratedTGNLayer_70557722738859
// MI455X (gfx1250) — compile-verified, matrix operand order checked by reading, not run on hardware
//
#include <hip/hip_runtime.h>
#include <hip/hip_bf16.h>
#include <math.h>

// ---------------------------------------------------------------------------
// TGN layer for MI455X (gfx1250): all GEMMs via v_wmma_f32_16x16x32_bf16,
// fp32 accumulation, bf16 activations staged in LDS (double-buffered in the
// generic GEMM), fp32 weights converted to bf16 with the hardware cvt
// (weights ~12MB -> L2-resident).
// ---------------------------------------------------------------------------

typedef __bf16        v16bf __attribute__((ext_vector_type(16)));
typedef float         v8f   __attribute__((ext_vector_type(8)));
typedef unsigned int  u32x4 __attribute__((ext_vector_type(4)));
typedef float         f32x4 __attribute__((ext_vector_type(4)));

#define DEV static __device__ __forceinline__

constexpr int B_EVT  = 65536;
constexpr int MDIM   = 512;
constexpr int TDIM   = 128;
constexpr int K1     = 2 * MDIM + 512 + TDIM;   // 1664
constexpr int NNODES = 200000;

// hardware f32 -> bf16 convert (RNE); compiler emits v_cvt_*bf16_f32
DEV unsigned short f2bf(float f) {
  __bf16 h = (__bf16)f;
  return __builtin_bit_cast(unsigned short, h);
}

union FragAB { v16bf v; u32x4 q[2]; };
union FragC  { v8f   v; float f[8]; };
union Pack8  { unsigned short s[8]; u32x4 q; };

// Load one 16x32 bf16 operand fragment from an LDS tile stored row-major
// [row][32].  Per ISA 16-bit A layout: lane L holds row M=L%16; lanes<16 get
// K={0..7,16..23}, lanes>=16 get K={8..15,24..31}  (two 16B ds reads).
DEV v16bf load_frag(const unsigned short* lds, int rowBase, int lane) {
  const int m  = lane & 15;
  const int h8 = (lane >> 4) << 3;
  const unsigned short* p = lds + (rowBase + m) * 32;
  FragAB f;
  f.q[0] = *(const u32x4*)(p + h8);
  f.q[1] = *(const u32x4*)(p + 16 + h8);
  return f.v;
}

DEV v8f wmma_bf16(v16bf a, v16bf b, v8f c) {
  return __builtin_amdgcn_wmma_f32_16x16x32_bf16(false, a, false, b,
                                                 (short)0, c, false, false);
}

DEV float sigmoidf_(float x) { return 1.0f / (1.0f + expf(-x)); }

// ---------------------------------------------------------------------------
// Kernel 1: new_memory = memory  (full bank copy, float4)
// ---------------------------------------------------------------------------
__global__ __launch_bounds__(256) void copy_mem_kernel(const f32x4* __restrict__ src,
                                                       f32x4* __restrict__ dst) {
  size_t i = (size_t)blockIdx.x * 256 + threadIdx.x;
  dst[i] = src[i];
}

// ---------------------------------------------------------------------------
// Kernel 2: gather memories -> bf16, time encoding -> bf16
// ---------------------------------------------------------------------------
__global__ __launch_bounds__(256) void prep_kernel(
    const float* __restrict__ memory, const int* __restrict__ src_ids,
    const int* __restrict__ dst_ids, const float* __restrict__ ts,
    const float* __restrict__ tw, const float* __restrict__ tb,
    unsigned short* __restrict__ srcA, unsigned short* __restrict__ dstA,
    unsigned short* __restrict__ temb) {
  const int b = blockIdx.x;
  const int t = threadIdx.x;
  const size_t sid = (size_t)src_ids[b], did = (size_t)dst_ids[b];
  for (int e = t; e < MDIM; e += 256) {
    srcA[(size_t)b * MDIM + e] = f2bf(memory[sid * MDIM + e]);
    dstA[(size_t)b * MDIM + e] = f2bf(memory[did * MDIM + e]);
  }
  if (t < TDIM) {
    float tv = ts[b];
    temb[(size_t)b * TDIM + t] = f2bf(cosf(tv * tw[t] + tb[t]));
  }
}

// ---------------------------------------------------------------------------
// Generic segmented-A GEMM:  C[B x 512] = act(A[B x K] @ W[512 x K]^T + bias)
// A is a row-wise concat of up to 4 segments (bf16 buffers or fp32 sources).
// Tile 128x128, 256 threads = 8 waves, each wave 32x64 (2x4 fragments).
// Double-buffered LDS: global loads for chunk k+1 overlap WMMAs of chunk k.
// MODE 0: ReLU -> bf16 ; MODE 1: identity -> bf16 ; MODE 2: identity -> fp32.
// ---------------------------------------------------------------------------
struct ASegs {
  const void* ptr[4];
  int start[5];   // cumulative K starts; start[4] == K
  int dtype[4];   // 0 = bf16 (ushort), 1 = fp32
  int width[4];   // per-segment row width in elements
};

template <int MODE>
__global__ __launch_bounds__(256) void gemm_kernel(ASegs segs,
    const float* __restrict__ W, const float* __restrict__ bias, int K,
    void* outp, int rowOff) {
  __shared__ unsigned short lA[2][128 * 32];
  __shared__ unsigned short lB[2][128 * 32];
  const int tid  = threadIdx.x;
  const int lane = tid & 31;
  const int wave = tid >> 5;
  const int wr = wave & 3;          // row group (32 rows)
  const int wc = wave >> 2;         // col group (64 cols)
  const int rows0 = blockIdx.y * 128;
  const int cols0 = blockIdx.x * 128;

  FragC acc[2][4];
  #pragma unroll
  for (int i = 0; i < 2; ++i)
    #pragma unroll
    for (int j = 0; j < 4; ++j)
      #pragma unroll
      for (int r = 0; r < 8; ++r) acc[i][j].f[r] = 0.0f;

  const int lrow = tid >> 1;        // 0..127
  const int lkk  = (tid & 1) << 4;  // 0 or 16

  // stage global -> regs for one 32-wide K chunk
  auto loadA = [&](int k0, u32x4& r0, u32x4& r1) {
    int s = 0;
    while (k0 >= segs.start[s + 1]) ++s;            // uniform (scalar) search
    const int kl = k0 - segs.start[s] + lkk;
    if (segs.dtype[s] == 0) {
      const unsigned short* p = (const unsigned short*)segs.ptr[s] +
          (size_t)(rows0 + lrow) * segs.width[s] + kl;
      r0 = *(const u32x4*)p;
      r1 = *(const u32x4*)(p + 8);
    } else {
      const float* p = (const float*)segs.ptr[s] +
          (size_t)(rows0 + lrow) * segs.width[s] + kl;
      f32x4 f0 = *(const f32x4*)p;
      f32x4 f1 = *(const f32x4*)(p + 4);
      f32x4 f2 = *(const f32x4*)(p + 8);
      f32x4 f3 = *(const f32x4*)(p + 12);
      Pack8 p0, p1;
      #pragma unroll
      for (int i = 0; i < 4; ++i) {
        p0.s[i] = f2bf(f0[i]); p0.s[4 + i] = f2bf(f1[i]);
        p1.s[i] = f2bf(f2[i]); p1.s[4 + i] = f2bf(f3[i]);
      }
      r0 = p0.q; r1 = p1.q;
    }
  };
  auto loadB = [&](int k0, u32x4& r0, u32x4& r1) {
    const float* p = W + (size_t)(cols0 + lrow) * K + k0 + lkk;
    f32x4 f0 = *(const f32x4*)p;
    f32x4 f1 = *(const f32x4*)(p + 4);
    f32x4 f2 = *(const f32x4*)(p + 8);
    f32x4 f3 = *(const f32x4*)(p + 12);
    Pack8 p0, p1;
    #pragma unroll
    for (int i = 0; i < 4; ++i) {
      p0.s[i] = f2bf(f0[i]); p0.s[4 + i] = f2bf(f1[i]);
      p1.s[i] = f2bf(f2[i]); p1.s[4 + i] = f2bf(f3[i]);
    }
    r0 = p0.q; r1 = p1.q;
  };
  auto stash = [&](int buf, u32x4 a0, u32x4 a1, u32x4 b0, u32x4 b1) {
    *(u32x4*)(&lA[buf][lrow * 32 + lkk])     = a0;
    *(u32x4*)(&lA[buf][lrow * 32 + lkk + 8]) = a1;
    *(u32x4*)(&lB[buf][lrow * 32 + lkk])     = b0;
    *(u32x4*)(&lB[buf][lrow * 32 + lkk + 8]) = b1;
  };

  {   // prologue: fill buffer 0
    u32x4 a0, a1, b0, b1;
    loadA(0, a0, a1);
    loadB(0, b0, b1);
    stash(0, a0, a1, b0, b1);
  }
  __syncthreads();

  int cur = 0;
  for (int k0 = 0; k0 < K; k0 += 32) {
    const bool nx = (k0 + 32) < K;
    u32x4 na0, na1, nb0, nb1;
    if (nx) { loadA(k0 + 32, na0, na1); loadB(k0 + 32, nb0, nb1); }

    v16bf fa0 = load_frag(&lA[cur][0], wr * 32, lane);
    v16bf fa1 = load_frag(&lA[cur][0], wr * 32 + 16, lane);
    #pragma unroll
    for (int j = 0; j < 4; ++j) {
      v16bf fb = load_frag(&lB[cur][0], wc * 64 + j * 16, lane);
      acc[0][j].v = wmma_bf16(fa0, fb, acc[0][j].v);
      acc[1][j].v = wmma_bf16(fa1, fb, acc[1][j].v);
    }

    if (nx) stash(cur ^ 1, na0, na1, nb0, nb1);
    __syncthreads();
    cur ^= 1;
  }

  // epilogue: C layout -> VGPR r holds M=r (lanes<16) / M=r+8 (lanes>=16), N=lane%16
  #pragma unroll
  for (int i = 0; i < 2; ++i)
    #pragma unroll
    for (int j = 0; j < 4; ++j)
      #pragma unroll
      for (int r = 0; r < 8; ++r) {
        const int row = rows0 + wr * 32 + i * 16 + r + ((lane >> 4) << 3);
        const int col = cols0 + wc * 64 + j * 16 + (lane & 15);
        float v = acc[i][j].f[r] + bias[col];
        if (MODE == 0) {
          v = fmaxf(v, 0.0f);
          ((unsigned short*)outp)[(size_t)row * 512 + col] = f2bf(v);
        } else if (MODE == 1) {
          ((unsigned short*)outp)[(size_t)row * 512 + col] = f2bf(v);
        } else {
          ((float*)outp)[(size_t)(rowOff + row) * 512 + col] = v;
        }
      }
}

// ---------------------------------------------------------------------------
// Fused GRU cell: six simultaneous WMMA GEMMs (gi_r/z/n from message x,
// gh_r/z/n from hidden h), gate math in fp32, scatter upd -> new_memory[ids],
// plus bf16 copy of upd for the output projection.  Tile 64x64, 8 waves,
// each wave 16x32 per gate-part -> 12 accumulator fragments.  Weights are
// L2-resident (re-read by all 1024 row tiles), so single-buffered LDS plus
// 8192 concurrent workgroups already covers the load latency.
// ---------------------------------------------------------------------------
__global__ __launch_bounds__(256) void gru_kernel(
    const unsigned short* __restrict__ xbf, const unsigned short* __restrict__ hbf,
    const float* __restrict__ w_ih, const float* __restrict__ w_hh,
    const float* __restrict__ b_ih, const float* __restrict__ b_hh,
    const float* __restrict__ memory, const int* __restrict__ ids,
    float* __restrict__ newmem, unsigned short* __restrict__ updbf) {
  __shared__ unsigned short lX[64 * 32];
  __shared__ unsigned short lH[64 * 32];
  __shared__ unsigned short lW[6][64 * 32];
  const int tid  = threadIdx.x;
  const int lane = tid & 31;
  const int wave = tid >> 5;
  const int wr = wave & 3;          // 16-row group
  const int wc = wave >> 2;         // 32-col group
  const int rows0 = blockIdx.y * 64;
  const int cols0 = blockIdx.x * 64;

  FragC acc[6][2];
  #pragma unroll
  for (int wt = 0; wt < 6; ++wt)
    #pragma unroll
    for (int cf = 0; cf < 2; ++cf)
      #pragma unroll
      for (int r = 0; r < 8; ++r) acc[wt][cf].f[r] = 0.0f;

  const int lrow = tid >> 2;        // 0..63
  const int lkk  = (tid & 3) << 3;  // 0,8,16,24

  for (int k0 = 0; k0 < 512; k0 += 32) {
    {
      const unsigned short* px = xbf + (size_t)(rows0 + lrow) * 512 + k0 + lkk;
      const unsigned short* ph = hbf + (size_t)(rows0 + lrow) * 512 + k0 + lkk;
      *(u32x4*)(&lX[lrow * 32 + lkk]) = *(const u32x4*)px;
      *(u32x4*)(&lH[lrow * 32 + lkk]) = *(const u32x4*)ph;
    }
    #pragma unroll
    for (int wt = 0; wt < 6; ++wt) {
      const float* Wm = (wt < 3) ? w_ih : w_hh;
      const int g = (wt < 3) ? wt : wt - 3;
      const float* p = Wm + (size_t)(g * 512 + cols0 + lrow) * 512 + k0 + lkk;
      f32x4 f0 = *(const f32x4*)p;
      f32x4 f1 = *(const f32x4*)(p + 4);
      Pack8 pk;
      #pragma unroll
      for (int i = 0; i < 4; ++i) { pk.s[i] = f2bf(f0[i]); pk.s[4 + i] = f2bf(f1[i]); }
      *(u32x4*)(&lW[wt][lrow * 32 + lkk]) = pk.q;
    }
    __syncthreads();
    v16bf ax = load_frag(lX, wr * 16, lane);
    v16bf ah = load_frag(lH, wr * 16, lane);
    #pragma unroll
    for (int wt = 0; wt < 6; ++wt) {
      v16bf a = (wt < 3) ? ax : ah;
      #pragma unroll
      for (int cf = 0; cf < 2; ++cf) {
        v16bf b = load_frag(lW[wt], wc * 32 + cf * 16, lane);
        acc[wt][cf].v = wmma_bf16(a, b, acc[wt][cf].v);
      }
    }
    __syncthreads();
  }

  #pragma unroll
  for (int cf = 0; cf < 2; ++cf)
    #pragma unroll
    for (int r = 0; r < 8; ++r) {
      const int row = rows0 + wr * 16 + r + ((lane >> 4) << 3);
      const int col = cols0 + wc * 32 + cf * 16 + (lane & 15);
      float gr  = acc[0][cf].f[r] + acc[3][cf].f[r] + b_ih[col] + b_hh[col];
      float gz  = acc[1][cf].f[r] + acc[4][cf].f[r] + b_ih[512 + col] + b_hh[512 + col];
      float inn = acc[2][cf].f[r] + b_ih[1024 + col];
      float hnn = acc[5][cf].f[r] + b_hh[1024 + col];
      float rg = sigmoidf_(gr);
      float zg = sigmoidf_(gz);
      float nn = tanhf(inn + rg * hnn);
      size_t id = (size_t)ids[row];
      float h   = memory[id * 512 + col];
      float upd = (1.0f - zg) * nn + zg * h;
      newmem[id * 512 + col] = upd;
      updbf[(size_t)row * 512 + col] = f2bf(upd);
    }
}

// ---------------------------------------------------------------------------
extern "C" void kernel_launch(void* const* d_in, const int* in_sizes, int n_in,
                              void* d_out, int out_size, void* d_ws, size_t ws_size,
                              hipStream_t stream) {
  const float* src_emb  = (const float*)d_in[0];
  const float* dst_emb  = (const float*)d_in[1];
  const float* edge     = (const float*)d_in[2];
  const float* ts       = (const float*)d_in[3];
  const float* memory   = (const float*)d_in[4];
  const float* time_w   = (const float*)d_in[5];
  const float* time_b   = (const float*)d_in[6];
  const float* msg_w1   = (const float*)d_in[7];
  const float* msg_b1   = (const float*)d_in[8];
  const float* msg_w2   = (const float*)d_in[9];
  const float* msg_b2   = (const float*)d_in[10];
  const float* gru_w_ih = (const float*)d_in[11];
  const float* gru_w_hh = (const float*)d_in[12];
  const float* gru_b_ih = (const float*)d_in[13];
  const float* gru_b_hh = (const float*)d_in[14];
  const float* out_w    = (const float*)d_in[15];
  const float* out_b    = (const float*)d_in[16];
  const int*   src_ids  = (const int*)d_in[17];
  const int*   dst_ids  = (const int*)d_in[18];

  float* out_f  = (float*)d_out;
  float* newmem = out_f + (size_t)2 * B_EVT * MDIM;   // new_memory region

  char* w = (char*)d_ws;
  const size_t SZBF = (size_t)B_EVT * 512 * sizeof(unsigned short);
  unsigned short* srcA = (unsigned short*)w; w += SZBF;
  unsigned short* dstA = (unsigned short*)w; w += SZBF;
  unsigned short* temb = (unsigned short*)w; w += (size_t)B_EVT * TDIM * 2;
  unsigned short* h1f  = (unsigned short*)w; w += SZBF;
  unsigned short* h1b  = (unsigned short*)w; w += SZBF;
  unsigned short* msgf = (unsigned short*)w; w += SZBF;
  unsigned short* msgb = (unsigned short*)w; w += SZBF;
  unsigned short* updsrc = h1f;   // h1f dead after msg layer-2 fwd
  unsigned short* upddst = h1b;   // h1b dead after msg layer-2 bwd

  // 1. new_memory = memory
  copy_mem_kernel<<<(NNODES * 512) / 4 / 256, 256, 0, stream>>>(
      (const f32x4*)memory, (f32x4*)newmem);

  // 2. gather + time encoding
  prep_kernel<<<B_EVT, 256, 0, stream>>>(memory, src_ids, dst_ids, ts,
                                         time_w, time_b, srcA, dstA, temb);

  dim3 g512(4, B_EVT / 128);

  // 3/4. message MLP layer 1 (K=1664, ReLU)
  ASegs sf{};
  sf.ptr[0] = srcA; sf.ptr[1] = dstA; sf.ptr[2] = edge; sf.ptr[3] = temb;
  sf.start[0] = 0; sf.start[1] = 512; sf.start[2] = 1024; sf.start[3] = 1536; sf.start[4] = K1;
  sf.dtype[0] = 0; sf.dtype[1] = 0; sf.dtype[2] = 1; sf.dtype[3] = 0;
  sf.width[0] = 512; sf.width[1] = 512; sf.width[2] = 512; sf.width[3] = 128;
  gemm_kernel<0><<<g512, 256, 0, stream>>>(sf, msg_w1, msg_b1, K1, h1f, 0);

  ASegs sb = sf; sb.ptr[0] = dstA; sb.ptr[1] = srcA;
  gemm_kernel<0><<<g512, 256, 0, stream>>>(sb, msg_w1, msg_b1, K1, h1b, 0);

  // 5/6. message MLP layer 2 (K=512)
  ASegs s2{};
  s2.ptr[0] = h1f;
  s2.start[0] = 0; s2.start[1] = 512; s2.start[2] = 512; s2.start[3] = 512; s2.start[4] = 512;
  s2.dtype[0] = 0; s2.width[0] = 512;
  gemm_kernel<1><<<g512, 256, 0, stream>>>(s2, msg_w2, msg_b2, 512, msgf, 0);  // src_to_dst
  ASegs s2b = s2; s2b.ptr[0] = h1b;
  gemm_kernel<1><<<g512, 256, 0, stream>>>(s2b, msg_w2, msg_b2, 512, msgb, 0); // dst_to_src

  // 7/8. fused GRU + scatter into new_memory
  dim3 ggru(512 / 64, B_EVT / 64);
  gru_kernel<<<ggru, 256, 0, stream>>>(msgb, srcA, gru_w_ih, gru_w_hh,
      gru_b_ih, gru_b_hh, memory, src_ids, newmem, updsrc);
  gru_kernel<<<ggru, 256, 0, stream>>>(msgf, dstA, gru_w_ih, gru_w_hh,
      gru_b_ih, gru_b_hh, memory, dst_ids, newmem, upddst);

  // 9/10. output projection (K=1024), fp32 results into d_out rows [0,2B)
  ASegs so{};
  so.ptr[0] = updsrc; so.ptr[1] = src_emb;
  so.start[0] = 0; so.start[1] = 512; so.start[2] = 1024; so.start[3] = 1024; so.start[4] = 1024;
  so.dtype[0] = 0; so.dtype[1] = 1; so.width[0] = 512; so.width[1] = 512;
  gemm_kernel<2><<<g512, 256, 0, stream>>>(so, out_w, out_b, 1024, out_f, 0);
  ASegs so2 = so; so2.ptr[0] = upddst; so2.ptr[1] = dst_emb;
  gemm_kernel<2><<<g512, 256, 0, stream>>>(so2, out_w, out_b, 1024, out_f, B_EVT);
}
